// CaptioningRNN_82240033784241
// MI455X (gfx1250) — matrix operand order repaired and checked by reading, not compiled, not run on hardware
//
#include <hip/hip_runtime.h>
#include <hip/hip_bf16.h>
#include <math.h>

// ---------------------------------------------------------------------------
// Attention-LSTM for MI455X (gfx1250): bf16 WMMA GEMMs + fused gate epilogue.
// N=128, T=64, D=1024, H=1024, 4H=4096, K2=2048 (concat [h, attn]).
// Software-pipelined K-loops: next-iteration fragments prefetched before the
// current iteration's WMMAs so v_wmma issues without draining loadcnt.
// ---------------------------------------------------------------------------

typedef __attribute__((ext_vector_type(16))) __bf16 v16bf;
typedef __attribute__((ext_vector_type(8)))  float  v8f;

#define NB   128
#define TT   64
#define DD   1024
#define HH   1024
#define FOURH 4096
#define KCAT 2048

union FragU { v16bf v; uint4 q[2]; };

__device__ __forceinline__ __bf16 f2bf(float f) {
  union { float f; unsigned u; } in; in.f = f;
  unsigned u = in.u + 0x7FFFu + ((in.u >> 16) & 1u);
  union { unsigned short s; __bf16 b; } out; out.s = (unsigned short)(u >> 16);
  return out.b;
}

__device__ __forceinline__ float sigmoidf(float x) {
  return 1.0f / (1.0f + __expf(-x));
}

// A-fragment: 16x32 bf16, row-major source (stride elems).
// lane L: M = L&15, half = L>>4; VGPR0..3 = K(half*8 .. +7), VGPR4..7 = K(16+half*8 .. +7)
__device__ __forceinline__ v16bf load_a_frag(const __bf16* base, int stride, int lane) {
  const int m = lane & 15, half = (lane >> 4) & 1;
  const __bf16* p = base + (size_t)m * stride + half * 8;
  FragU f;
  f.q[0] = *(const uint4*)(p);
  f.q[1] = *(const uint4*)(p + 16);
  return f.v;
}

// B-fragment: 32x16 bf16 taken from B^T (Ncols x K row-major, stride elems).
// lane L: col = L&15, half = L>>4; VGPR0..3 = K(half*8 .. +7), VGPR4..7 = K(16+half*8 .. +7)
__device__ __forceinline__ v16bf load_b_frag(const __bf16* baseT, int stride, int lane) {
  const int n = lane & 15, half = (lane >> 4) & 1;
  const __bf16* p = baseT + (size_t)n * stride + half * 8;
  FragU f;
  f.q[0] = *(const uint4*)(p);
  f.q[1] = *(const uint4*)(p + 16);
  return f.v;
}

__device__ __forceinline__ v8f wmma_bf16(v16bf a, v16bf b, v8f c) {
  return __builtin_amdgcn_wmma_f32_16x16x32_bf16(false, a, false, b, (short)0, c, false, false);
}

// --------------------------- prologue kernels ------------------------------

__global__ void k_conv_bf16(const float* __restrict__ src, __bf16* __restrict__ dst, int n) {
  int i = blockIdx.x * blockDim.x + threadIdx.x;
  if (i < n) dst[i] = f2bf(src[i]);
}

// WxT[c, k] = Wx[k, c]   (4096 x 1024 bf16)
__global__ void k_make_wxT(const float* __restrict__ Wx, __bf16* __restrict__ wxT) {
  int i = blockIdx.x * blockDim.x + threadIdx.x;   // over 4096*1024
  int k = i & (DD - 1);
  int c = i >> 10;
  wxT[i] = f2bf(Wx[(size_t)k * FOURH + c]);
}

// WcatT[c, k] = (k<H ? Wh[k,c] : Wattn[k-H,c])   (4096 x 2048 bf16)
__global__ void k_make_wcatT(const float* __restrict__ Wh, const float* __restrict__ Wattn,
                             __bf16* __restrict__ wcatT) {
  int i = blockIdx.x * blockDim.x + threadIdx.x;   // over 4096*2048
  int k = i & (KCAT - 1);
  int c = i >> 11;
  float v = (k < HH) ? Wh[(size_t)k * FOURH + c]
                     : Wattn[(size_t)(k - HH) * FOURH + c];
  wcatT[i] = f2bf(v);
}

// h0 = c0 = mean over the 16 spatial cells of Aflat (N,H,16)
__global__ void k_init_h(const float* __restrict__ A, float* __restrict__ hbuf,
                         float* __restrict__ cbuf) {
  int i = blockIdx.x * blockDim.x + threadIdx.x;   // over N*H
  if (i >= NB * HH) return;
  const float* ap = A + (size_t)i * 16;
  float s = 0.f;
  #pragma unroll
  for (int m = 0; m < 16; ++m) s += ap[m];
  s *= (1.0f / 16.0f);
  hbuf[i] = s;
  cbuf[i] = s;
}

// --------------------- xW = x @ Wx  (stored as (T, N, 4H)) -----------------
// grid (512, 16), block 128 (4 waves); each wave computes a 16x64 strip:
// one A-fragment reused across 4 B-tiles per K step, double-buffered.
__global__ void k_gemm_xw(const __bf16* __restrict__ xb, const __bf16* __restrict__ wxT,
                          float* __restrict__ xw_all) {
  const int lane = threadIdx.x & 31;
  const int wid  = threadIdx.x >> 5;
  const int row0 = blockIdx.x * 16;                       // row = n*T + t
  const int c0   = (blockIdx.y * 4 + wid) * 64;           // 64-col strip

  const __bf16* abase = xb  + (size_t)row0 * DD;
  const __bf16* bb0 = wxT + (size_t)(c0 +  0) * DD;
  const __bf16* bb1 = wxT + (size_t)(c0 + 16) * DD;
  const __bf16* bb2 = wxT + (size_t)(c0 + 32) * DD;
  const __bf16* bb3 = wxT + (size_t)(c0 + 48) * DD;

  v8f acc0 = {}, acc1 = {}, acc2 = {}, acc3 = {};

  v16bf af = load_a_frag(abase, DD, lane);
  v16bf f0 = load_b_frag(bb0, DD, lane);
  v16bf f1 = load_b_frag(bb1, DD, lane);
  v16bf f2 = load_b_frag(bb2, DD, lane);
  v16bf f3 = load_b_frag(bb3, DD, lane);

  #pragma unroll 2
  for (int k = 0; k < DD - 32; k += 32) {
    // prefetch next K-slice before consuming the current one
    v16bf afn = load_a_frag(abase + k + 32, DD, lane);
    v16bf g0  = load_b_frag(bb0 + k + 32, DD, lane);
    v16bf g1  = load_b_frag(bb1 + k + 32, DD, lane);
    v16bf g2  = load_b_frag(bb2 + k + 32, DD, lane);
    v16bf g3  = load_b_frag(bb3 + k + 32, DD, lane);
    acc0 = wmma_bf16(af, f0, acc0);
    acc1 = wmma_bf16(af, f1, acc1);
    acc2 = wmma_bf16(af, f2, acc2);
    acc3 = wmma_bf16(af, f3, acc3);
    af = afn; f0 = g0; f1 = g1; f2 = g2; f3 = g3;
  }
  acc0 = wmma_bf16(af, f0, acc0);
  acc1 = wmma_bf16(af, f1, acc1);
  acc2 = wmma_bf16(af, f2, acc2);
  acc3 = wmma_bf16(af, f3, acc3);

  const int nl = lane & 15;
  #pragma unroll
  for (int r = 0; r < 8; ++r) {
    int m   = (lane < 16) ? r : (r + 8);
    int row = row0 + m;
    int nb  = row >> 6;        // row / T
    int t   = row & (TT - 1);  // row % T
    float* orow = xw_all + ((size_t)t * NB + nb) * FOURH;
    orow[c0 +  0 + nl] = acc0[r];
    orow[c0 + 16 + nl] = acc1[r];
    orow[c0 + 32 + nl] = acc2[r];
    orow[c0 + 48 + nl] = acc3[r];
  }
}

// --------------------------- per-step attention ----------------------------
// 1 block per batch n, 256 threads. Emits bf16 Z[n] = [h | attn] (2048 wide).
__global__ void k_attn(const float* __restrict__ A, const float* __restrict__ hbuf,
                       __bf16* __restrict__ Z) {
  __shared__ float red[256][16];
  const int n   = blockIdx.x;
  const int tid = threadIdx.x;
  const float* An = A + (size_t)n * HH * 16;
  const float* hn = hbuf + (size_t)n * HH;

  float part[16];
  #pragma unroll
  for (int m = 0; m < 16; ++m) part[m] = 0.f;

  for (int h = tid; h < HH; h += 256) {
    float hv = hn[h];
    const float* ap = An + (size_t)h * 16;
    #pragma unroll
    for (int m = 0; m < 16; ++m) part[m] += hv * ap[m];
  }
  #pragma unroll
  for (int m = 0; m < 16; ++m) red[tid][m] = part[m];
  __syncthreads();
  for (int s = 128; s > 0; s >>= 1) {
    if (tid < s) {
      #pragma unroll
      for (int m = 0; m < 16; ++m) red[tid][m] += red[tid + s][m];
    }
    __syncthreads();
  }

  // softmax over 16 (computed redundantly by every thread)
  const float scale = 0.03125f;  // 1/sqrt(1024)
  float sc[16], mx = -3.0e38f;
  #pragma unroll
  for (int m = 0; m < 16; ++m) { sc[m] = red[0][m] * scale; mx = fmaxf(mx, sc[m]); }
  float sum = 0.f;
  #pragma unroll
  for (int m = 0; m < 16; ++m) { sc[m] = __expf(sc[m] - mx); sum += sc[m]; }
  float inv = 1.0f / sum;
  #pragma unroll
  for (int m = 0; m < 16; ++m) sc[m] *= inv;

  __bf16* zn = Z + (size_t)n * KCAT;
  for (int h = tid; h < HH; h += 256) {
    const float* ap = An + (size_t)h * 16;
    float acc = 0.f;
    #pragma unroll
    for (int m = 0; m < 16; ++m) acc += ap[m] * sc[m];
    zn[h]      = f2bf(hn[h]);   // h part (K 0..1023)
    zn[HH + h] = f2bf(acc);     // attn part (K 1024..2047)
  }
}

// ------------- fused recurrent GEMM + LSTM gates (one timestep) ------------
// act = Z @ Wcat + xw_t + b ; gates fused in the WMMA accumulator layout.
// grid (8, 16), block 128 (4 waves). Each wave: 16 rows x 16 cols x 4 gates,
// double-buffered K-loop.
__global__ void k_step(const __bf16* __restrict__ Z, const __bf16* __restrict__ wcatT,
                       const float* __restrict__ xw_t, const float* __restrict__ bias,
                       float* __restrict__ cbuf, float* __restrict__ hbuf,
                       float* __restrict__ out, int t) {
  const int lane = threadIdx.x & 31;
  const int wid  = threadIdx.x >> 5;
  const int row0 = blockIdx.x * 16;                 // batch rows
  const int c0   = (blockIdx.y * 4 + wid) * 16;     // 0..1008 within H

  const __bf16* abase = Z + (size_t)row0 * KCAT;
  const __bf16* b0 = wcatT + (size_t)(0 * HH + c0) * KCAT;
  const __bf16* b1 = wcatT + (size_t)(1 * HH + c0) * KCAT;
  const __bf16* b2 = wcatT + (size_t)(2 * HH + c0) * KCAT;
  const __bf16* b3 = wcatT + (size_t)(3 * HH + c0) * KCAT;

  v8f acc0 = {}, acc1 = {}, acc2 = {}, acc3 = {};

  v16bf af = load_a_frag(abase, KCAT, lane);
  v16bf f0 = load_b_frag(b0, KCAT, lane);
  v16bf f1 = load_b_frag(b1, KCAT, lane);
  v16bf f2 = load_b_frag(b2, KCAT, lane);
  v16bf f3 = load_b_frag(b3, KCAT, lane);

  #pragma unroll 2
  for (int k = 0; k < KCAT - 32; k += 32) {
    v16bf afn = load_a_frag(abase + k + 32, KCAT, lane);
    v16bf g0  = load_b_frag(b0 + k + 32, KCAT, lane);
    v16bf g1  = load_b_frag(b1 + k + 32, KCAT, lane);
    v16bf g2  = load_b_frag(b2 + k + 32, KCAT, lane);
    v16bf g3  = load_b_frag(b3 + k + 32, KCAT, lane);
    acc0 = wmma_bf16(af, f0, acc0);
    acc1 = wmma_bf16(af, f1, acc1);
    acc2 = wmma_bf16(af, f2, acc2);
    acc3 = wmma_bf16(af, f3, acc3);
    af = afn; f0 = g0; f1 = g1; f2 = g2; f3 = g3;
  }
  acc0 = wmma_bf16(af, f0, acc0);
  acc1 = wmma_bf16(af, f1, acc1);
  acc2 = wmma_bf16(af, f2, acc2);
  acc3 = wmma_bf16(af, f3, acc3);

  const int col = c0 + (lane & 15);
  const float bi  = bias[col];
  const float bff = bias[HH + col];
  const float bo  = bias[2 * HH + col];
  const float bg  = bias[3 * HH + col];

  #pragma unroll
  for (int r = 0; r < 8; ++r) {
    int m   = (lane < 16) ? r : (r + 8);
    int row = row0 + m;
    const float* xw = xw_t + (size_t)row * FOURH;
    float iv = sigmoidf(acc0[r] + xw[col]          + bi);
    float fv = sigmoidf(acc1[r] + xw[HH + col]     + bff);
    float ov = sigmoidf(acc2[r] + xw[2 * HH + col] + bo);
    float gv = tanhf   (acc3[r] + xw[3 * HH + col] + bg);
    size_t hc = (size_t)row * HH + col;
    float cn = fv * cbuf[hc] + iv * gv;
    float hn = ov * tanhf(cn);
    cbuf[hc] = cn;
    hbuf[hc] = hn;
    out[((size_t)row * TT + t) * HH + col] = hn;
  }
}

// --------------------------------- launch ----------------------------------

extern "C" void kernel_launch(void* const* d_in, const int* in_sizes, int n_in,
                              void* d_out, int out_size, void* d_ws, size_t ws_size,
                              hipStream_t stream) {
  const float* x     = (const float*)d_in[0];   // (N,T,D)
  const float* A     = (const float*)d_in[1];   // (N,H,4,4) == (N,H,16)
  const float* Wx    = (const float*)d_in[2];   // (D,4H)
  const float* Wh    = (const float*)d_in[3];   // (H,4H)
  const float* Wattn = (const float*)d_in[4];   // (H,4H)
  const float* b     = (const float*)d_in[5];   // (4H)
  float* out = (float*)d_out;                   // (N,T,H)

  char* ws = (char*)d_ws;
  size_t off = 0;
  auto take = [&](size_t bytes) { char* p = ws + off; off = (off + bytes + 255) & ~(size_t)255; return p; };

  float*  xw_all = (float*)take((size_t)TT * NB * FOURH * 4);   // 134 MB, (T,N,4H)
  __bf16* xb     = (__bf16*)take((size_t)NB * TT * DD * 2);     // 16.8 MB
  __bf16* wxT    = (__bf16*)take((size_t)FOURH * DD * 2);       // 8.4 MB
  __bf16* wcatT  = (__bf16*)take((size_t)FOURH * KCAT * 2);     // 16.8 MB
  __bf16* Zb     = (__bf16*)take((size_t)NB * KCAT * 2);        // 0.5 MB
  float*  hbuf   = (float*)take((size_t)NB * HH * 4);
  float*  cbuf   = (float*)take((size_t)NB * HH * 4);
  (void)ws_size; (void)in_sizes; (void)n_in; (void)out_size;

  // Prologue: conversions, transposed weights, initial state.
  {
    int n = NB * TT * DD;
    k_conv_bf16<<<(n + 255) / 256, 256, 0, stream>>>(x, xb, n);
  }
  k_make_wxT<<<(FOURH * DD) / 256, 256, 0, stream>>>(Wx, wxT);
  k_make_wcatT<<<(FOURH * KCAT) / 256, 256, 0, stream>>>(Wh, Wattn, wcatT);
  k_init_h<<<(NB * HH) / 256, 256, 0, stream>>>(A, hbuf, cbuf);

  // Big input-projection GEMM, written as (T, N, 4H).
  k_gemm_xw<<<dim3((NB * TT) / 16, FOURH / 256), 128, 0, stream>>>(xb, wxT, xw_all);

  // 64 serial recurrent steps.
  for (int t = 0; t < TT; ++t) {
    k_attn<<<NB, 256, 0, stream>>>(A, hbuf, Zb);
    const float* xw_t = xw_all + (size_t)t * NB * FOURH;
    k_step<<<dim3(NB / 16, HH / 64), 128, 0, stream>>>(Zb, wcatT, xw_t, b,
                                                       cbuf, hbuf, out, t);
  }
}
